// FrontierPolicyNetwork_16716012716419
// MI455X (gfx1250) — compile-verified, matrix operand order checked by reading, not run on hardware
//
#include <hip/hip_runtime.h>
#include <hip/hip_bf16.h>

typedef __attribute__((ext_vector_type(16))) _Float16 v16h;
typedef __attribute__((ext_vector_type(8)))  float    v8f;

#define NN 50000
#define EE 800000
#define FF 64
#define HH 128
#define CC 20000
#define PP 200000
#define NFF 2048
#define LL 3

// K-index mapping for 16-bit A/B WMMA fragments (ISA 7.12.2, 16-bit A 16x32):
// lane half g=0: V0..3 -> K 0..7, V4..7 -> K 16..23 ; g=1: +8 / +8.
__device__ __forceinline__ int kmap(int i, int g) {
    int v = i >> 1, p = i & 1;
    return (v < 4) ? (8 * g + 2 * v + p) : (16 + 8 * g + 2 * (v - 4) + p);
}
// even base K of VGPR pair v
__device__ __forceinline__ int kpair(int v, int g) {
    return (v < 4) ? (8 * g + 2 * v) : (16 + 8 * g + 2 * (v - 4));
}

__device__ __forceinline__ v8f wmma_f16(v16h a, v16h b, v8f c) {
    return __builtin_amdgcn_wmma_f32_16x16x32_f16(false, a, false, b, (short)0, c, false, false);
}

union FragU { v16h h; unsigned int u[8]; };

// Build A fragment from an LDS f16 tile (row stride in halves, must keep k 4B-aligned).
template <int STRIDE>
__device__ __forceinline__ v16h load_a_frag(const _Float16* lds, int mr, int g, int k0) {
    FragU f;
#pragma unroll
    for (int v = 0; v < 8; ++v) {
        int k = k0 + kpair(v, g);
        f.u[v] = *(const unsigned int*)(lds + mr * STRIDE + k);
    }
    return f.h;
}

// ---------------- utility kernels ----------------
__global__ void zero_f32(float* __restrict__ p, int n) {
    for (int i = blockIdx.x * blockDim.x + threadIdx.x; i < n; i += gridDim.x * blockDim.x)
        p[i] = 0.f;
}

// Swizzle f32 weight [K][128] into fragment-major f16:
// dst[(((t*KB + kb)*32 + lane)*16) + i] = W[(kb*32 + kmap(i, lane>>4))*128 + t*16 + (lane&15)]
// so each (wave,kb) B-fragment is one contiguous 32-byte load.
__global__ void swz_w16(const float* __restrict__ W, _Float16* __restrict__ dst, int KB, int total) {
    for (int idx = blockIdx.x * blockDim.x + threadIdx.x; idx < total; idx += gridDim.x * blockDim.x) {
        int i = idx & 15;
        int rest = idx >> 4;
        int lane = rest & 31; rest >>= 5;
        int kb = rest % KB;
        int t  = rest / KB;
        int g = lane >> 4, nc = lane & 15;
        int k = kb * 32 + kmap(i, g);
        dst[idx] = (_Float16)W[(size_t)k * HH + t * 16 + nc];
    }
}

// ---------------- input projection: x = nf @ in_W + in_b ----------------
__global__ void input_proj(const float* __restrict__ nf, const _Float16* __restrict__ Wh,
                           const float* __restrict__ bias, float* __restrict__ x) {
    __shared__ _Float16 Asub[16][72];
    const int m0 = blockIdx.x * 16, tid = threadIdx.x;
    for (int idx = tid; idx < 16 * 64; idx += 256) {
        int r = idx >> 6, c = idx & 63;
        Asub[r][c] = (_Float16)nf[(size_t)(m0 + r) * FF + c];
    }
    __syncthreads();
    const int wv = tid >> 5, lane = tid & 31;
    const int n0 = wv * 16, g = lane >> 4, mr = lane & 15, nc = lane & 15;
    v8f acc = {};
#pragma unroll
    for (int kb = 0; kb < 2; ++kb) {
        v16h af = load_a_frag<72>(&Asub[0][0], mr, g, kb * 32);
        v16h bf = *(const v16h*)(Wh + ((size_t)(wv * 2 + kb) * 32 + lane) * 16);
        acc = wmma_f16(af, bf, acc);
    }
    const float b = bias[n0 + nc];
#pragma unroll
    for (int v = 0; v < 8; ++v)
        x[(size_t)(m0 + v + 8 * g) * HH + n0 + nc] = acc[v] + b;
}

// ---------------- edge message: agg[dst] += relu(x[src] + ea*eW + eb) ----------------
__global__ void edge_msg(const float* __restrict__ x, const int* __restrict__ ei,
                         const float* __restrict__ ea, const float* __restrict__ eW,
                         const float* __restrict__ eb, float* __restrict__ agg, int e_) {
    const int lane = threadIdx.x & 31;
    const int wave = blockIdx.x * (blockDim.x >> 5) + (threadIdx.x >> 5);
    const int nwv  = gridDim.x * (blockDim.x >> 5);
    const int f = lane * 4;
    const float4 w4 = *(const float4*)(eW + f);
    const float4 b4 = *(const float4*)(eb + f);
    for (int j = wave; j < e_; j += nwv) {
        const int s = ei[j], d = ei[e_ + j];
        const float a = ea[j];
        const float4 xv = *(const float4*)(x + (size_t)s * HH + f);
        float m0 = fmaxf(xv.x + a * w4.x + b4.x, 0.f);
        float m1 = fmaxf(xv.y + a * w4.y + b4.y, 0.f);
        float m2 = fmaxf(xv.z + a * w4.z + b4.z, 0.f);
        float m3 = fmaxf(xv.w + a * w4.w + b4.w, 0.f);
        float* dp = agg + (size_t)d * HH + f;
        atomicAdd(dp + 0, m0);
        atomicAdd(dp + 1, m1);
        atomicAdd(dp + 2, m2);
        atomicAdd(dp + 3, m3);
    }
}

// ---------------- GINE node MLP: xout = relu(relu((agg+x)@w1+b1)@w2+b2) ----------------
__global__ void node_mlp(const float* __restrict__ x, const float* __restrict__ agg,
                         const _Float16* __restrict__ w1, const float* __restrict__ b1,
                         const _Float16* __restrict__ w2, const float* __restrict__ b2,
                         float* __restrict__ xout) {
    __shared__ _Float16 Asub[16][136];
    __shared__ _Float16 Hsub[16][136];
    const int m0 = blockIdx.x * 16, tid = threadIdx.x;
    for (int idx = tid; idx < 16 * 128; idx += 256) {
        int r = idx >> 7, c = idx & 127;
        size_t o = (size_t)(m0 + r) * HH + c;
        Asub[r][c] = (_Float16)(agg[o] + x[o]);
    }
    __syncthreads();
    const int wv = tid >> 5, lane = tid & 31;
    const int n0 = wv * 16, g = lane >> 4, mr = lane & 15, nc = lane & 15;
    v8f acc = {};
#pragma unroll
    for (int kb = 0; kb < 4; ++kb) {
        v16h af = load_a_frag<136>(&Asub[0][0], mr, g, kb * 32);
        v16h bf = *(const v16h*)(w1 + ((size_t)(wv * 4 + kb) * 32 + lane) * 16);
        acc = wmma_f16(af, bf, acc);
    }
    const float bb1 = b1[n0 + nc];
#pragma unroll
    for (int v = 0; v < 8; ++v)
        Hsub[v + 8 * g][n0 + nc] = (_Float16)fmaxf(acc[v] + bb1, 0.f);
    __syncthreads();
    v8f acc2 = {};
#pragma unroll
    for (int kb = 0; kb < 4; ++kb) {
        v16h af = load_a_frag<136>(&Hsub[0][0], mr, g, kb * 32);
        v16h bf = *(const v16h*)(w2 + ((size_t)(wv * 4 + kb) * 32 + lane) * 16);
        acc2 = wmma_f16(af, bf, acc2);
    }
    const float bb2 = b2[n0 + nc];
#pragma unroll
    for (int v = 0; v < 8; ++v)
        xout[(size_t)(m0 + v + 8 * g) * HH + n0 + nc] = fmaxf(acc2[v] + bb2, 0.f);
}

// ---------------- pool scatter: zsum[mem[p]] += x[idx[p]], zcnt[mem[p]]++ ----------------
__global__ void pool_scatter(const float* __restrict__ x, const int* __restrict__ pidx,
                             const int* __restrict__ pmem, float* __restrict__ zsum,
                             float* __restrict__ zcnt, int p_) {
    const int lane = threadIdx.x & 31;
    const int wave = blockIdx.x * (blockDim.x >> 5) + (threadIdx.x >> 5);
    const int nwv  = gridDim.x * (blockDim.x >> 5);
    const int f = lane * 4;
    for (int p = wave; p < p_; p += nwv) {
        const int node = pidx[p], c = pmem[p];
        const float4 xv = *(const float4*)(x + (size_t)node * HH + f);
        float* dp = zsum + (size_t)c * HH + f;
        atomicAdd(dp + 0, xv.x);
        atomicAdd(dp + 1, xv.y);
        atomicAdd(dp + 2, xv.z);
        atomicAdd(dp + 3, xv.w);
        if (lane == 0) atomicAdd(zcnt + c, 1.f);
    }
}

// ---------------- z = zsum/max(cnt,1) (in place) ; fsum[cb[c]] += z ; fcnt++ ----------------
__global__ void z_finalize(float* __restrict__ z, const float* __restrict__ zcnt,
                           const int* __restrict__ cb, float* __restrict__ fsum,
                           float* __restrict__ fcnt, int c_) {
    const int lane = threadIdx.x & 31;
    const int wave = blockIdx.x * (blockDim.x >> 5) + (threadIdx.x >> 5);
    const int nwv  = gridDim.x * (blockDim.x >> 5);
    const int f = lane * 4;
    for (int c = wave; c < c_; c += nwv) {
        const float inv = 1.f / fmaxf(zcnt[c], 1.f);
        const int b = cb[c];
        float* zp = z + (size_t)c * HH + f;
        float4 v = *(const float4*)zp;
        v.x *= inv; v.y *= inv; v.z *= inv; v.w *= inv;
        *(float4*)zp = v;
        float* fp = fsum + (size_t)b * HH + f;
        atomicAdd(fp + 0, v.x);
        atomicAdd(fp + 1, v.y);
        atomicAdd(fp + 2, v.z);
        atomicAdd(fp + 3, v.w);
        if (lane == 0) atomicAdd(fcnt + b, 1.f);
    }
}

// ---------------- head: logits = mlp([z|ctx]) with mask ----------------
__global__ void head(const float* __restrict__ z, const int* __restrict__ cb,
                     const float* __restrict__ fsum, const float* __restrict__ fcnt,
                     const _Float16* __restrict__ w0, const float* __restrict__ b0,
                     const _Float16* __restrict__ w1, const float* __restrict__ b1,
                     const float* __restrict__ w2, const float* __restrict__ b2,
                     const unsigned char* __restrict__ mask, float* __restrict__ out) {
    __shared__ _Float16 Asub[16][264];
    __shared__ _Float16 Hsub[16][136];
    __shared__ _Float16 Gsub[16][136];
    const int c0 = blockIdx.x * 16, tid = threadIdx.x;
    for (int idx = tid; idx < 16 * 256; idx += 256) {
        int r = idx >> 8, col = idx & 255;
        int c = c0 + r;
        float val;
        if (col < 128) {
            val = z[(size_t)c * HH + col];
        } else {
            int b = cb[c];
            val = fsum[(size_t)b * HH + (col - 128)] / fmaxf(fcnt[b], 1.f);
        }
        Asub[r][col] = (_Float16)val;
    }
    __syncthreads();
    const int wv = tid >> 5, lane = tid & 31;
    const int n0 = wv * 16, g = lane >> 4, mr = lane & 15, nc = lane & 15;
    // layer 0: [16,256]@[256,128]
    v8f acc = {};
#pragma unroll
    for (int kb = 0; kb < 8; ++kb) {
        v16h af = load_a_frag<264>(&Asub[0][0], mr, g, kb * 32);
        v16h bf = *(const v16h*)(w0 + ((size_t)(wv * 8 + kb) * 32 + lane) * 16);
        acc = wmma_f16(af, bf, acc);
    }
    const float bb0 = b0[n0 + nc];
#pragma unroll
    for (int v = 0; v < 8; ++v)
        Hsub[v + 8 * g][n0 + nc] = (_Float16)fmaxf(acc[v] + bb0, 0.f);
    __syncthreads();
    // layer 1: [16,128]@[128,128]
    v8f acc2 = {};
#pragma unroll
    for (int kb = 0; kb < 4; ++kb) {
        v16h af = load_a_frag<136>(&Hsub[0][0], mr, g, kb * 32);
        v16h bf = *(const v16h*)(w1 + ((size_t)(wv * 4 + kb) * 32 + lane) * 16);
        acc2 = wmma_f16(af, bf, acc2);
    }
    const float bb1 = b1[n0 + nc];
#pragma unroll
    for (int v = 0; v < 8; ++v)
        Gsub[v + 8 * g][n0 + nc] = (_Float16)fmaxf(acc2[v] + bb1, 0.f);
    __syncthreads();
    // final [128,1] dot per row; wave wv handles rows wv and wv+8
#pragma unroll
    for (int rr = 0; rr < 2; ++rr) {
        int r = wv + 8 * rr;
        float s = 0.f;
        for (int f = lane; f < 128; f += 32) s += (float)Gsub[r][f] * w2[f];
#pragma unroll
        for (int off = 16; off; off >>= 1) s += __shfl_down(s, off);
        if (lane == 0) {
            int c = c0 + r;
            float lg = s + b2[0];
            out[c] = mask[c] ? lg : -1.0e9f;
        }
    }
}

extern "C" void kernel_launch(void* const* d_in, const int* in_sizes, int n_in,
                              void* d_out, int out_size, void* d_ws, size_t ws_size,
                              hipStream_t stream) {
    const float* nodef = (const float*)d_in[0];
    const int*   ei    = (const int*)d_in[1];
    const float* ea    = (const float*)d_in[2];
    const int*   cb    = (const int*)d_in[4];
    const unsigned char* mask = (const unsigned char*)d_in[5];
    const int*   pidx  = (const int*)d_in[6];
    const int*   pmem  = (const int*)d_in[7];
    const float* in_W  = (const float*)d_in[8];
    const float* in_b  = (const float*)d_in[9];
    const float* e_W   = (const float*)d_in[10];
    const float* e_b   = (const float*)d_in[11];
    const float* cw1   = (const float*)d_in[12];
    const float* cb1   = (const float*)d_in[13];
    const float* cw2   = (const float*)d_in[14];
    const float* cb2   = (const float*)d_in[15];
    const float* h0W   = (const float*)d_in[16];
    const float* h0b   = (const float*)d_in[17];
    const float* h1W   = (const float*)d_in[18];
    const float* h1b   = (const float*)d_in[19];
    const float* h2W   = (const float*)d_in[20];
    const float* h2b   = (const float*)d_in[21];
    float* out = (float*)d_out;

    const int e_ = in_sizes[1] / 2;   // 800000
    const int c_ = in_sizes[4];       // 20000
    const int p_ = in_sizes[6];       // 200000

    // ---- workspace layout ----
    char* base = (char*)d_ws;
    size_t off = 0;
    auto alloc = [&](size_t bytes) -> char* {
        char* p = base + off;
        off = (off + bytes + 255) & ~(size_t)255;
        return p;
    };
    float* x0  = (float*)alloc((size_t)NN * HH * 4);
    float* x1  = (float*)alloc((size_t)NN * HH * 4);
    float* agg = (float*)alloc((size_t)NN * HH * 4);
    // contiguous pool region zeroed in one launch: zsum | fsum | zcnt | fcnt
    char* poolreg = alloc((size_t)CC * HH * 4 + (size_t)NFF * HH * 4 + (size_t)CC * 4 + (size_t)NFF * 4);
    float* zsum = (float*)poolreg;
    float* fsum = zsum + (size_t)CC * HH;
    float* zcnt = fsum + (size_t)NFF * HH;
    float* fcnt = zcnt + CC;
    const int pool_floats = CC * HH + NFF * HH + CC + NFF;
    _Float16* inWh = (_Float16*)alloc((size_t)FF * HH * 2);
    _Float16* w1h  = (_Float16*)alloc((size_t)LL * HH * HH * 2);
    _Float16* w2h  = (_Float16*)alloc((size_t)LL * HH * HH * 2);
    _Float16* h0h  = (_Float16*)alloc((size_t)2 * HH * HH * 2);
    _Float16* h1h  = (_Float16*)alloc((size_t)HH * HH * 2);

    // ---- weight conversion + fragment swizzle (f32 -> f16, fragment-major) ----
    swz_w16<<<64, 256, 0, stream>>>(in_W, inWh, 2, FF * HH);
    for (int l = 0; l < LL; ++l) {
        swz_w16<<<64, 256, 0, stream>>>(cw1 + (size_t)l * HH * HH, w1h + (size_t)l * HH * HH, 4, HH * HH);
        swz_w16<<<64, 256, 0, stream>>>(cw2 + (size_t)l * HH * HH, w2h + (size_t)l * HH * HH, 4, HH * HH);
    }
    swz_w16<<<128, 256, 0, stream>>>(h0W, h0h, 8, 2 * HH * HH);
    swz_w16<<<64, 256, 0, stream>>>(h1W, h1h, 4, HH * HH);

    // ---- input projection ----
    input_proj<<<NN / 16, 256, 0, stream>>>(nodef, inWh, in_b, x0);

    // ---- GINE layers ----
    float* xc = x0;
    float* xn = x1;
    for (int l = 0; l < LL; ++l) {
        zero_f32<<<4096, 256, 0, stream>>>(agg, NN * HH);
        edge_msg<<<4096, 256, 0, stream>>>(xc, ei, ea, e_W, e_b, agg, e_);
        node_mlp<<<NN / 16, 256, 0, stream>>>(xc, agg,
                                              w1h + (size_t)l * HH * HH, cb1 + (size_t)l * HH,
                                              w2h + (size_t)l * HH * HH, cb2 + (size_t)l * HH,
                                              xn);
        float* t = xc; xc = xn; xn = t;
    }

    // ---- candidate pooling + frontier context ----
    zero_f32<<<4096, 256, 0, stream>>>(zsum, pool_floats);
    pool_scatter<<<2048, 256, 0, stream>>>(xc, pidx, pmem, zsum, zcnt, p_);
    z_finalize<<<1024, 256, 0, stream>>>(zsum, zcnt, cb, fsum, fcnt, c_);

    // ---- head MLP ----
    head<<<c_ / 16, 256, 0, stream>>>(zsum, cb, fsum, fcnt,
                                      h0h, h0b, h1h, h1b, h2W, h2b, mask, out);
}
